// MultiHeadAttention_52347061404126
// MI455X (gfx1250) — compile-verified, matrix operand order checked by reading, not run on hardware
//
#include <hip/hip_runtime.h>
#include <hip/hip_bf16.h>
#include <cstdint>

// ---------------------------------------------------------------------------
// MI455X (gfx1250) pre-norm MHA block, bf16 WMMA pipeline, f32 accumulate.
// Fragments load as 2x b128; inner loops register-double-buffered; attention
// K-blocks staged into LDS via the Tensor Data Mover when available.
// ---------------------------------------------------------------------------

typedef __attribute__((ext_vector_type(16))) __bf16 v16bf;
typedef __attribute__((ext_vector_type(8)))  float  v8f;
typedef __attribute__((ext_vector_type(4)))  unsigned u32x4;
typedef __attribute__((ext_vector_type(8)))  int      i32x8;
typedef __attribute__((ext_vector_type(4)))  int      i32x4;

#define WMMA_BF16(a, b, c) \
  __builtin_amdgcn_wmma_f32_16x16x32_bf16(false, (a), false, (b), (short)0, (c), false, false)

#if defined(__has_builtin)
#if __has_builtin(__builtin_amdgcn_tensor_load_to_lds) && \
    __has_builtin(__builtin_amdgcn_s_wait_tensorcnt)
#define HAVE_TDM 1
#endif
#endif
#ifndef HAVE_TDM
#define HAVE_TDM 0
#endif

static constexpr int B_  = 4;
static constexpr int S_  = 2048;
static constexpr int D_  = 1024;
static constexpr int H_  = 16;
static constexpr int HD_ = 64;

union FragA { v16bf v; uint4 q[2]; __bf16 e[16]; };
union Acc   { v8f   v; float    f[8]; };

// --- 16-bit A-style fragment (16x32): rows on lane%16, K per ISA 7.12.2.
// Elements 0..7 = K(half*8 + 0..7), elements 8..15 = K(half*8 + 16..23):
// two 16-byte contiguous chunks -> two b128 loads (global or ds).
__device__ inline v16bf load_frag_a16(const __bf16* __restrict__ p, int ld) {
  const int lane = threadIdx.x & 31;
  const int ln = lane & 15, lh = lane >> 4;
  FragA f;
  const __bf16* row = p + (size_t)ln * ld + lh * 8;
  f.q[0] = *reinterpret_cast<const uint4*>(row);
  f.q[1] = *reinterpret_cast<const uint4*>(row + 16);
  return f.v;
}

#if HAVE_TDM
// --- TDM: async-load a dense 2D bf16 tile (32 rows x 64 cols, row stride 64)
// from global into LDS.  D# per CDNA5 ISA.
// All inputs are wave-uniform.  Tracked by TENSORcnt.
__device__ inline void tdm_load_k_block(const __bf16* gsrc, __bf16* lds_dst) {
  const unsigned long long ga = (unsigned long long)(uintptr_t)gsrc;
  const unsigned lds_off = (unsigned)(uintptr_t)lds_dst;  // flat LDS: addr[31:0]
  u32x4 g0;
  g0[0] = 1u;                                        // count=1, user D#
  g0[1] = lds_off;                                   // lds_addr (bytes)
  g0[2] = (unsigned)(ga & 0xFFFFFFFFu);              // global_addr[31:0]
  g0[3] = (unsigned)((ga >> 32) & 0x01FFFFFFu) | (2u << 30);  // [56:32] | type=2
  i32x8 g1;
  g1[0] = 1 << 16;             // workgroup_mask=0, data_size=1 (2 bytes)
  g1[1] = (int)(64u << 16);    // tensor_dim0 = 64 (low16 at bits 63:48)
  g1[2] = (int)(2048u << 16);  // tensor_dim0 hi=0 | tensor_dim1 = 2048 (low16)
  g1[3] = (int)(64u << 16);    // tensor_dim1 hi=0 | tile_dim0 = 64
  g1[4] = 32;                  // tile_dim1 = 32, tile_dim2 = 0
  g1[5] = 64;                  // tensor_dim0_stride = 64 (elements)
  g1[6] = 0;                   // stride hi | tensor_dim1_stride lo
  g1[7] = 0;
  const i32x4 z4 = {0, 0, 0, 0};
#if __clang_major__ >= 23
  const i32x8 z8 = {0, 0, 0, 0, 0, 0, 0, 0};
  __builtin_amdgcn_tensor_load_to_lds(g0, g1, z4, z4, z8, 0);
#else
  __builtin_amdgcn_tensor_load_to_lds(g0, g1, z4, z4, 0);
#endif
}
#endif  // HAVE_TDM

// ---------------------------------------------------------------------------
// 1) LayerNorm (eps=1e-5) + cast to bf16.  One 256-thread block per row.
// ---------------------------------------------------------------------------
__global__ void ln_bf16_kernel(const float* __restrict__ x,
                               const float* __restrict__ gamma,
                               const float* __restrict__ beta,
                               __bf16* __restrict__ xn) {
  const int row = blockIdx.x;          // B*S rows
  const int tid = threadIdx.x;         // 256 threads, 4 elems each
  const float4 v = reinterpret_cast<const float4*>(x + (size_t)row * D_)[tid];
  __shared__ float rs_[256], rss_[256];
  rs_[tid]  = v.x + v.y + v.z + v.w;
  rss_[tid] = v.x * v.x + v.y * v.y + v.z * v.z + v.w * v.w;
  __syncthreads();
  for (int off = 128; off > 0; off >>= 1) {
    if (tid < off) { rs_[tid] += rs_[tid + off]; rss_[tid] += rss_[tid + off]; }
    __syncthreads();
  }
  const float mu  = rs_[0]  * (1.0f / D_);
  const float var = rss_[0] * (1.0f / D_) - mu * mu;
  const float inv = rsqrtf(var + 1e-5f);
  const int c = tid * 4;
  __bf16* o = xn + (size_t)row * D_ + c;
  o[0] = (__bf16)((v.x - mu) * inv * gamma[c + 0] + beta[c + 0]);
  o[1] = (__bf16)((v.y - mu) * inv * gamma[c + 1] + beta[c + 1]);
  o[2] = (__bf16)((v.z - mu) * inv * gamma[c + 2] + beta[c + 2]);
  o[3] = (__bf16)((v.w - mu) * inv * gamma[c + 3] + beta[c + 3]);
}

// ---------------------------------------------------------------------------
// 2) Transpose + f32->bf16: dst[n*K + k] = (bf16)src[k*N + n].
// ---------------------------------------------------------------------------
__global__ void transpose_f32_to_bf16_kernel(const float* __restrict__ src,
                                             __bf16* __restrict__ dst,
                                             int K, int N) {
  __shared__ float tile[32][33];
  const int n0 = blockIdx.x * 32, k0 = blockIdx.y * 32;
  const int tx = threadIdx.x, ty = threadIdx.y;     // block (32, 8)
  for (int i = ty; i < 32; i += 8)
    tile[i][tx] = src[(size_t)(k0 + i) * N + n0 + tx];
  __syncthreads();
  for (int i = ty; i < 32; i += 8)
    dst[(size_t)(n0 + i) * K + k0 + tx] = (__bf16)tile[tx][i];
}

// ---------------------------------------------------------------------------
// GEMM core: wave tile 32x64, K-step 32, register double buffering.
// ---------------------------------------------------------------------------
__device__ inline void gemm_core_32x64(const __bf16* __restrict__ A, int m0,
                                       const __bf16* __restrict__ WT, int n0,
                                       Acc acc[2][4], int lane) {
  const __bf16* ar0 = A + (size_t)m0 * D_;
  const __bf16* ar1 = A + (size_t)(m0 + 16) * D_;
  const __bf16* wr0 = WT + (size_t)(n0 + 0)  * D_;
  const __bf16* wr1 = WT + (size_t)(n0 + 16) * D_;
  const __bf16* wr2 = WT + (size_t)(n0 + 32) * D_;
  const __bf16* wr3 = WT + (size_t)(n0 + 48) * D_;

  v16bf a0 = load_frag_a16(ar0, D_);
  v16bf a1 = load_frag_a16(ar1, D_);
  v16bf b0 = load_frag_a16(wr0, D_);
  v16bf b1 = load_frag_a16(wr1, D_);
  v16bf b2 = load_frag_a16(wr2, D_);
  v16bf b3 = load_frag_a16(wr3, D_);

  for (int k0 = 0; k0 < D_; k0 += 32) {
    const int kn = (k0 + 32 < D_) ? k0 + 32 : k0;   // clamp: last iter reloads
    if (k0 + 64 < D_) {                              // prefetch k+2
      __builtin_prefetch(ar0 + (size_t)(lane & 15) * D_ + k0 + 64, 0, 3);
      __builtin_prefetch(wr0 + (size_t)lane * D_ + k0 + 64, 0, 3);
      __builtin_prefetch(wr2 + (size_t)lane * D_ + k0 + 64, 0, 3);
    }
    // next step's loads in fresh regs -> overlap with WMMAs below
    const v16bf na0 = load_frag_a16(ar0 + kn, D_);
    const v16bf na1 = load_frag_a16(ar1 + kn, D_);
    const v16bf nb0 = load_frag_a16(wr0 + kn, D_);
    const v16bf nb1 = load_frag_a16(wr1 + kn, D_);
    const v16bf nb2 = load_frag_a16(wr2 + kn, D_);
    const v16bf nb3 = load_frag_a16(wr3 + kn, D_);

    acc[0][0].v = WMMA_BF16(a0, b0, acc[0][0].v);
    acc[1][0].v = WMMA_BF16(a1, b0, acc[1][0].v);
    acc[0][1].v = WMMA_BF16(a0, b1, acc[0][1].v);
    acc[1][1].v = WMMA_BF16(a1, b1, acc[1][1].v);
    acc[0][2].v = WMMA_BF16(a0, b2, acc[0][2].v);
    acc[1][2].v = WMMA_BF16(a1, b2, acc[1][2].v);
    acc[0][3].v = WMMA_BF16(a0, b3, acc[0][3].v);
    acc[1][3].v = WMMA_BF16(a1, b3, acc[1][3].v);

    a0 = na0; a1 = na1; b0 = nb0; b1 = nb1; b2 = nb2; b3 = nb3;
  }
}

// ---------------------------------------------------------------------------
// 3) QKV GEMM -> Q/K [B,H,S,hd] and V transposed [B,H,hd,S] (all bf16).
// ---------------------------------------------------------------------------
__global__ void qkv_gemm_kernel(const __bf16* __restrict__ A,
                                const __bf16* __restrict__ WT,   // [3N x K]
                                const float* __restrict__ bias,
                                __bf16* __restrict__ Qo,
                                __bf16* __restrict__ Ko,
                                __bf16* __restrict__ VTo) {
  const int lane = threadIdx.x & 31;
  const int wave = threadIdx.x >> 5;
  const int ln = lane & 15, lh = lane >> 4;
  const int m0 = blockIdx.y * 64 + (wave >> 1) * 32;
  const int n0 = blockIdx.x * 128 + (wave & 1) * 64;

  Acc acc[2][4];
#pragma unroll
  for (int i = 0; i < 2; ++i)
#pragma unroll
    for (int j = 0; j < 4; ++j)
#pragma unroll
      for (int r = 0; r < 8; ++r) acc[i][j].f[r] = 0.0f;

  gemm_core_32x64(A, m0, WT, n0, acc, lane);

#pragma unroll
  for (int mt = 0; mt < 2; ++mt)
#pragma unroll
    for (int nt = 0; nt < 4; ++nt)
#pragma unroll
      for (int r = 0; r < 8; ++r) {
        const int row = m0 + mt * 16 + r + 8 * lh;       // C layout: M = r + 8*half
        const int col = n0 + nt * 16 + ln;               // N = lane % 16
        const float val = acc[mt][nt].f[r] + bias[col];
        const int which = col >> 10, dcol = col & 1023;
        const int hh = dcol >> 6, dd = dcol & 63;
        const int bb = row >> 11, ss = row & 2047;
        if (which == 2) {            // V stored transposed: [B,H,hd,S]
          VTo[(((size_t)bb * H_ + hh) * HD_ + dd) * S_ + ss] = (__bf16)val;
        } else {
          __bf16* dst = (which == 0) ? Qo : Ko;
          dst[(((size_t)bb * H_ + hh) * S_ + ss) * HD_ + dd] = (__bf16)val;
        }
      }
}

// ---------------------------------------------------------------------------
// 4) Flash attention: one wave per (b,h, 16-query tile).  Per 32-key step:
//    4 WMMAs Q*K^T, online softmax, P via LDS layout swap, 4 WMMAs P*V.
//    K blocks: TDM async staging into double-buffered LDS when available,
//    otherwise register double buffering.  V fragments issued early.
// ---------------------------------------------------------------------------
__global__ void attn_kernel(const __bf16* __restrict__ Q,
                            const __bf16* __restrict__ K,
                            const __bf16* __restrict__ VT,   // [B,H,hd,S]
                            __bf16* __restrict__ ctx) {
  const int bh = blockIdx.y;
  const int b = bh >> 4, h = bh & 15;
  const int q0 = blockIdx.x * 16;
  const int lane = threadIdx.x & 31;
  const int ln = lane & 15, lh = lane >> 4;

  const __bf16* qb = Q  + ((size_t)bh * S_ + q0) * HD_;
  const __bf16* kb = K  + (size_t)bh * S_ * HD_;
  const __bf16* vt = VT + (size_t)bh * HD_ * S_;

  const v16bf qa0 = load_frag_a16(qb, HD_);
  const v16bf qa1 = load_frag_a16(qb + 32, HD_);

  Acc o[4];
  float mrun[8], lrun[8];
#pragma unroll
  for (int t = 0; t < 4; ++t)
#pragma unroll
    for (int r = 0; r < 8; ++r) o[t].f[r] = 0.0f;
#pragma unroll
  for (int r = 0; r < 8; ++r) { mrun[r] = -1e30f; lrun[r] = 0.0f; }

  __shared__ alignas(16) __bf16 lds_p[16 * 32];  // P tile (C-layout -> A-layout)

#if HAVE_TDM
  __shared__ alignas(16) __bf16 kstage[2][32 * HD_];  // double-buffered K tiles
  tdm_load_k_block(kb, &kstage[0][0]);                // prime buffer 0
#else
  v16bf kf0 = load_frag_a16(kb, HD_);
  v16bf kf1 = load_frag_a16(kb + 32, HD_);
  v16bf kf2 = load_frag_a16(kb + (size_t)16 * HD_, HD_);
  v16bf kf3 = load_frag_a16(kb + (size_t)16 * HD_ + 32, HD_);
#endif

  for (int kk = 0; kk < S_; kk += 32) {
    const int kn = (kk + 32 < S_) ? kk + 32 : kk;
#if HAVE_TDM
    const int cur = (kk >> 5) & 1;
    tdm_load_k_block(kb + (size_t)kn * HD_, &kstage[cur ^ 1][0]);  // next block
#endif

    // V fragments for CURRENT step, issued early (consumed after softmax)
    const v16bf vf0 = load_frag_a16(vt + (size_t)0  * S_ + kk, S_);
    const v16bf vf1 = load_frag_a16(vt + (size_t)16 * S_ + kk, S_);
    const v16bf vf2 = load_frag_a16(vt + (size_t)32 * S_ + kk, S_);
    const v16bf vf3 = load_frag_a16(vt + (size_t)48 * S_ + kk, S_);
#if !HAVE_TDM
    const v16bf nk0 = load_frag_a16(kb + (size_t)kn * HD_, HD_);
    const v16bf nk1 = load_frag_a16(kb + (size_t)kn * HD_ + 32, HD_);
    const v16bf nk2 = load_frag_a16(kb + (size_t)(kn + 16) * HD_, HD_);
    const v16bf nk3 = load_frag_a16(kb + (size_t)(kn + 16) * HD_ + 32, HD_);
#endif
    if (kk + 64 < S_) {
      __builtin_prefetch(kb + (size_t)(kk + 64 + lane) * HD_, 0, 3);
      __builtin_prefetch(vt + (size_t)lane * S_ + kk + 64, 0, 3);
      __builtin_prefetch(vt + (size_t)(32 + lane) * S_ + kk + 64, 0, 3);
    }

#if HAVE_TDM
    // wait for the CURRENT buffer's DMA (<=1 outstanding keeps next in flight)
    __builtin_amdgcn_s_wait_tensorcnt(1);
    asm volatile("" ::: "memory");
    const __bf16* kcur = &kstage[cur][0];
    const v16bf kf0 = load_frag_a16(kcur, HD_);                    // ds_load_b128
    const v16bf kf1 = load_frag_a16(kcur + 32, HD_);
    const v16bf kf2 = load_frag_a16(kcur + (size_t)16 * HD_, HD_);
    const v16bf kf3 = load_frag_a16(kcur + (size_t)16 * HD_ + 32, HD_);
#endif

    // ---- scores: two 16x16 tiles over hd=64 (two WMMAs each)
    Acc s0, s1;
#pragma unroll
    for (int r = 0; r < 8; ++r) { s0.f[r] = 0.0f; s1.f[r] = 0.0f; }
    s0.v = WMMA_BF16(qa0, kf0, s0.v);
    s0.v = WMMA_BF16(qa1, kf1, s0.v);
    s1.v = WMMA_BF16(qa0, kf2, s1.v);
    s1.v = WMMA_BF16(qa1, kf3, s1.v);

    // ---- online softmax over this 32-key block (row = r + 8*lh, col = ln)
    float alpha[8], psum[8];
#pragma unroll
    for (int r = 0; r < 8; ++r) {
      const float a0 = s0.f[r] * 0.125f;      // SCALE = hd^-0.5
      const float a1 = s1.f[r] * 0.125f;
      float m = fmaxf(a0, a1);
#pragma unroll
      for (int off = 1; off < 16; off <<= 1) m = fmaxf(m, __shfl_xor(m, off, 16));
      const float mnew = fmaxf(mrun[r], m);
      alpha[r] = __expf(mrun[r] - mnew);
      const float p0 = __expf(a0 - mnew);
      const float p1 = __expf(a1 - mnew);
      float ps = p0 + p1;
#pragma unroll
      for (int off = 1; off < 16; off <<= 1) ps += __shfl_xor(ps, off, 16);
      psum[r] = ps;
      mrun[r] = mnew;
      lds_p[(r + 8 * lh) * 32 + ln]      = (__bf16)p0;
      lds_p[(r + 8 * lh) * 32 + 16 + ln] = (__bf16)p1;
    }
    asm volatile("s_wait_dscnt 0" ::: "memory");   // LDS RAW across lanes
    const v16bf pa = load_frag_a16(&lds_p[0], 32); // 2x ds_load_b128

#pragma unroll
    for (int r = 0; r < 8; ++r) lrun[r] = lrun[r] * alpha[r] + psum[r];

    // ---- O = diag(alpha)*O + P x V
#pragma unroll
    for (int r = 0; r < 8; ++r) o[0].f[r] *= alpha[r];
    o[0].v = WMMA_BF16(pa, vf0, o[0].v);
#pragma unroll
    for (int r = 0; r < 8; ++r) o[1].f[r] *= alpha[r];
    o[1].v = WMMA_BF16(pa, vf1, o[1].v);
#pragma unroll
    for (int r = 0; r < 8; ++r) o[2].f[r] *= alpha[r];
    o[2].v = WMMA_BF16(pa, vf2, o[2].v);
#pragma unroll
    for (int r = 0; r < 8; ++r) o[3].f[r] *= alpha[r];
    o[3].v = WMMA_BF16(pa, vf3, o[3].v);

#if !HAVE_TDM
    kf0 = nk0; kf1 = nk1; kf2 = nk2; kf3 = nk3;
#endif
  }

  // ---- normalize by row sums and store ctx bf16 in [B,S,D] (D = h*64+d)
#pragma unroll
  for (int r = 0; r < 8; ++r) {
    const float inv = 1.0f / lrun[r];
    const int row = b * S_ + q0 + r + 8 * lh;
#pragma unroll
    for (int t = 0; t < 4; ++t)
      ctx[(size_t)row * D_ + h * HD_ + t * 16 + ln] = (__bf16)(o[t].f[r] * inv);
  }
}

// ---------------------------------------------------------------------------
// 5) Output projection: ctx[8192x1024] x w_projT[1024x1024](bf16) + b_proj
// ---------------------------------------------------------------------------
__global__ void proj_gemm_kernel(const __bf16* __restrict__ A,
                                 const __bf16* __restrict__ WT,  // [N x K]
                                 const float* __restrict__ bias,
                                 float* __restrict__ out) {
  const int lane = threadIdx.x & 31;
  const int wave = threadIdx.x >> 5;
  const int ln = lane & 15, lh = lane >> 4;
  const int m0 = blockIdx.y * 64 + (wave >> 1) * 32;
  const int n0 = blockIdx.x * 128 + (wave & 1) * 64;

  Acc acc[2][4];
#pragma unroll
  for (int i = 0; i < 2; ++i)
#pragma unroll
    for (int j = 0; j < 4; ++j)
#pragma unroll
      for (int r = 0; r < 8; ++r) acc[i][j].f[r] = 0.0f;

  gemm_core_32x64(A, m0, WT, n0, acc, lane);

#pragma unroll
  for (int mt = 0; mt < 2; ++mt)
#pragma unroll
    for (int nt = 0; nt < 4; ++nt)
#pragma unroll
      for (int r = 0; r < 8; ++r) {
        const int row = m0 + mt * 16 + r + 8 * lh;
        const int col = n0 + nt * 16 + ln;
        out[(size_t)row * D_ + col] = acc[mt][nt].f[r] + bias[col];
      }
}

// ---------------------------------------------------------------------------
// Launch
// ---------------------------------------------------------------------------
extern "C" void kernel_launch(void* const* d_in, const int* in_sizes, int n_in,
                              void* d_out, int out_size, void* d_ws, size_t ws_size,
                              hipStream_t stream) {
  const float* x      = (const float*)d_in[0];
  const float* gamma  = (const float*)d_in[1];
  const float* beta   = (const float*)d_in[2];
  const float* w_qkv  = (const float*)d_in[3];
  const float* b_qkv  = (const float*)d_in[4];
  const float* w_proj = (const float*)d_in[5];
  const float* b_proj = (const float*)d_in[6];
  float* out = (float*)d_out;

  char* base = (char*)d_ws;
  auto alloc = [&](size_t bytes) -> void* {
    void* p = (void*)base;
    base += (bytes + 255) & ~(size_t)255;
    return p;
  };

  const size_t MROWS = (size_t)B_ * S_;                        // 8192
  __bf16* xn      = (__bf16*)alloc(MROWS * D_ * 2);            // 16 MB
  __bf16* wqkvT   = (__bf16*)alloc((size_t)3 * D_ * D_ * 2);   //  6 MB  [3072 x 1024]
  __bf16* wprojT  = (__bf16*)alloc((size_t)D_ * D_ * 2);       //  2 MB  [1024 x 1024]
  __bf16* qb      = (__bf16*)alloc(MROWS * D_ * 2);            // 16 MB  [B,H,S,hd]
  __bf16* kb      = (__bf16*)alloc(MROWS * D_ * 2);            // 16 MB  [B,H,S,hd]
  __bf16* vtb     = (__bf16*)alloc(MROWS * D_ * 2);            // 16 MB  [B,H,hd,S]
  __bf16* cx      = (__bf16*)alloc(MROWS * D_ * 2);            // 16 MB  [B,S,D]

  ln_bf16_kernel<<<dim3((unsigned)MROWS), dim3(256), 0, stream>>>(x, gamma, beta, xn);

  transpose_f32_to_bf16_kernel<<<dim3(3 * D_ / 32, D_ / 32), dim3(32, 8), 0, stream>>>(
      w_qkv, wqkvT, D_, 3 * D_);
  transpose_f32_to_bf16_kernel<<<dim3(D_ / 32, D_ / 32), dim3(32, 8), 0, stream>>>(
      w_proj, wprojT, D_, D_);

  qkv_gemm_kernel<<<dim3(3 * D_ / 128, (unsigned)(MROWS / 64)), dim3(128), 0, stream>>>(
      xn, wqkvT, b_qkv, qb, kb, vtb);

  attn_kernel<<<dim3(S_ / 16, B_ * H_), dim3(32), 0, stream>>>(qb, kb, vtb, cx);

  proj_gemm_kernel<<<dim3(D_ / 128, (unsigned)(MROWS / 64)), dim3(128), 0, stream>>>(
      cx, wprojT, b_proj, out);
}
